// PreferencePropagation_68418829025317
// MI455X (gfx1250) — matrix operand order repaired and checked by reading, not compiled, not run on hardware
//
#include <hip/hip_runtime.h>
#include <math.h>

// ---------------------------------------------------------------------------
// RGCN (PyG-style) on MI455X / gfx1250.
//   layer(x) = relu( [mean-agg per (dst,rel) | x] @ [W_rel(1024x128); W_root] + b )
// Two layers, then sigmoid(h @ out_w + out_b).
// GEMMs: v_wmma_f32_16x16x32_bf16.
//   - A stored pre-swizzled in WMMA fragment-tiled layout (coalesced 32B/lane)
//   - B k-tile staged through LDS, ping-pong double-buffered (manual 2x unroll,
//     no register rotation copies)
//   - B fragments double-buffered in registers inside the nt loop
//   - 32-row M-tile per wave: each B fragment feeds two WMMAs
// Scatter: hardware global f32 atomics.
// ---------------------------------------------------------------------------

#define R_ 8
#define D_ 128
#define KFULL 1152          // R*D + D (root columns appended)
#define KTILES 36           // KFULL / 32 (even -> clean ping-pong)
#define ATILE_USH (KTILES * 32 * 16)   // ushorts per 16-row A tile (18432)

// Padded LDS B-tile: per fragment-lane slot 16 ushorts + (lane>>3)*8 pad.
#define BS_TILE_USH 4608

typedef __attribute__((ext_vector_type(16))) __bf16 v16bf;
typedef __attribute__((ext_vector_type(8)))  float  v8f;

union FragBF {
    v16bf v;
    uint4 u[2];
};

__device__ __forceinline__ unsigned short f32_to_bf16(float f) {
    unsigned int u = __float_as_uint(f);
    unsigned int r = u + 0x7FFFu + ((u >> 16) & 1u);   // round-to-nearest-even
    return (unsigned short)(r >> 16);
}

__device__ __forceinline__ void atomic_add_f32(float* p, float v) {
    __hip_atomic_fetch_add(p, v, __ATOMIC_RELAXED, __HIP_MEMORY_SCOPE_AGENT);
}

// -------------------------------- counting ---------------------------------

__global__ void rgcn_count_kernel(const int* __restrict__ ei,
                                  const int* __restrict__ et,
                                  unsigned int* __restrict__ cnt, int E) {
    int e = blockIdx.x * blockDim.x + threadIdx.x;
    if (e >= E) return;
    int dst = ei[E + e];
    int r   = et[e];
    atomicAdd(&cnt[(long long)dst * R_ + r], 1u);
}

__global__ void rgcn_inv_kernel(const unsigned int* __restrict__ cnt,
                                float* __restrict__ inv, int n) {
    int i = blockIdx.x * blockDim.x + threadIdx.x;
    if (i >= n) return;
    unsigned int c = cnt[i];
    inv[i] = 1.0f / (float)(c > 0u ? c : 1u);
}

// ------------------------------ edge scatter -------------------------------

__global__ void rgcn_scatter_kernel(const float* __restrict__ feat,
                                    const int* __restrict__ ei,
                                    const int* __restrict__ et,
                                    float* __restrict__ agg, int E) {
    long long gid = (long long)blockIdx.x * blockDim.x + threadIdx.x;
    int e    = (int)(gid >> 5);
    int lane = (int)(gid & 31);
    if (e >= E) return;
    int src = ei[e];
    int dst = ei[E + e];
    int r   = et[e];
    const float4 v = *(const float4*)(feat + (long long)src * D_ + lane * 4);
    float* base = agg + (((long long)dst * R_ + r) * D_) + lane * 4;
    atomic_add_f32(base + 0, v.x);
    atomic_add_f32(base + 1, v.y);
    atomic_add_f32(base + 2, v.z);
    atomic_add_f32(base + 3, v.w);
}

// ---------------- normalize + pack A in WMMA-fragment layout ---------------

__global__ void rgcn_build_a_kernel(const float* __restrict__ agg,
                                    const float* __restrict__ inv,
                                    const float* __restrict__ feat,
                                    unsigned short* __restrict__ A, int N) {
    long long gid = (long long)blockIdx.x * blockDim.x + threadIdx.x;
    int n  = (int)(gid / 144);
    int c8 = (int)(gid % 144);
    if (n >= N) return;
    int c = c8 * 8;
    float vals[8];
    if (c < R_ * D_) {
        int r = c >> 7;
        int d = c & (D_ - 1);
        float s = inv[(long long)n * R_ + r];
        const float* p = agg + (((long long)n * R_ + r) * D_) + d;
#pragma unroll
        for (int j = 0; j < 8; ++j) vals[j] = p[j] * s;
    } else {
        const float* p = feat + (long long)n * D_ + (c - R_ * D_);
#pragma unroll
        for (int j = 0; j < 8; ++j) vals[j] = p[j];
    }
    unsigned short o[8];
#pragma unroll
    for (int j = 0; j < 8; ++j) o[j] = f32_to_bf16(vals[j]);

    int kt   = c >> 5;
    int off  = c & 31;                        // 0, 8, 16, 24
    int lane = (n & 15) + ((off & 8) ? 16 : 0);
    int half = (off & 16) ? 1 : 0;
    size_t idx = (((size_t)(n >> 4) * KTILES + kt) * 32 + lane) * 16 + half * 8;
    *(uint4*)(A + idx) = *(const uint4*)o;
}

// ------------------------- pack B into WMMA layout -------------------------

__global__ void rgcn_pack_b_kernel(const float* __restrict__ Wrel,   // [1024,128]
                                   const float* __restrict__ Wroot,  // [128,128]
                                   unsigned short* __restrict__ Bp) {
    int t = blockIdx.x * blockDim.x + threadIdx.x;
    if (t >= KTILES * 8 * 32) return;
    int lane = t & 31;
    int f    = t >> 5;          // fragment index = kt*8 + nt
    int nt   = f & 7;
    int kt   = f >> 3;
    int col   = nt * 16 + (lane & 15);
    int kbase = kt * 32 + ((lane >> 4) ? 8 : 0);
    unsigned short o[16];
#pragma unroll
    for (int i = 0; i < 16; ++i) {
        int k = kbase + (i & 7) + ((i >> 3) << 4);
        float w = (k < R_ * D_) ? Wrel[(long long)k * D_ + col]
                                : Wroot[(long long)(k - R_ * D_) * D_ + col];
        o[i] = f32_to_bf16(w);
    }
    uint4* dst = (uint4*)(Bp + ((long long)f * 32 + lane) * 16);
    dst[0] = *(const uint4*)(o);
    dst[1] = *(const uint4*)(o + 8);
}

// ------------------------------- WMMA GEMM ---------------------------------

__device__ __forceinline__ int bs_off(int frag, int lane) {
    return (frag * 32 + lane) * 16 + (lane >> 3) * 8;   // ushort index, bank pad
}

__global__ void __launch_bounds__(256)
rgcn_gemm_kernel(const unsigned short* __restrict__ A,   // fragment-tiled
                 const unsigned short* __restrict__ Bp,
                 const float* __restrict__ bias,
                 float* __restrict__ out, int N) {
    __shared__ __align__(16) unsigned short Bs[2 * BS_TILE_USH];

    const int tid  = threadIdx.x;
    const int lane = tid & 31;
    const int wid  = tid >> 5;

    const int ntile16 = (N + 15) >> 4;
    int t0 = ((int)blockIdx.x * 8 + wid) * 2;
    if (t0 > ntile16 - 2) t0 = ntile16 - 2;   // clamp (duplicate, deterministic)
    const int t1 = t0 + 1;

    const unsigned short* a0p = A + (size_t)t0 * ATILE_USH + (size_t)lane * 16;
    const unsigned short* a1p = A + (size_t)t1 * ATILE_USH + (size_t)lane * 16;

    v8f acc0[8], acc1[8];
#pragma unroll
    for (int i = 0; i < 8; ++i) { acc0[i] = (v8f){}; acc1[i] = (v8f){}; }

    auto stageB = [&](int kt, int buf) {
        const uint4* src = (const uint4*)(Bp + ((size_t)(kt * 8 + wid) * 32 + lane) * 16);
        uint4 d0 = src[0];
        uint4 d1 = src[1];
        uint4* dst = (uint4*)(Bs + (size_t)buf * BS_TILE_USH + bs_off(wid, lane));
        dst[0] = d0;
        dst[1] = d1;
    };
    auto loadA = [&](const unsigned short* base, int kt, FragBF& f) {
        const uint4* s = (const uint4*)(base + (size_t)kt * 512);
        f.u[0] = s[0];
        f.u[1] = s[1];
    };
    // One B k-tile (8 fragments) from LDS base bl, register-pipelined, each
    // fragment feeding two WMMAs (rows t0, t1).
    auto mmaTile = [&](const unsigned short* bl, const FragBF& A0, const FragBF& A1) {
        FragBF bcur;
        {
            const uint4* q = (const uint4*)(bl + bs_off(0, lane));
            bcur.u[0] = q[0];
            bcur.u[1] = q[1];
        }
#pragma unroll
        for (int nt = 0; nt < 8; ++nt) {
            FragBF bnxt;
            if (nt < 7) {                   // prefetch next B frag before use
                const uint4* q = (const uint4*)(bl + bs_off(nt + 1, lane));
                bnxt.u[0] = q[0];
                bnxt.u[1] = q[1];
            }
            acc0[nt] = __builtin_amdgcn_wmma_f32_16x16x32_bf16(
                false, A0.v, false, bcur.v, (short)0, acc0[nt], false, false);
            acc1[nt] = __builtin_amdgcn_wmma_f32_16x16x32_bf16(
                false, A1.v, false, bcur.v, (short)0, acc1[nt], false, false);
            if (nt < 7) bcur = bnxt;        // renamed away by full unroll
        }
    };

    FragBF a0c, a1c, a0n, a1n;
    stageB(0, 0);
    loadA(a0p, 0, a0c);
    loadA(a1p, 0, a1c);
    __syncthreads();

    // Ping-pong over pairs of k-tiles: no fragment-register rotation copies.
#pragma clang loop unroll(disable)
    for (int kt = 0; kt < KTILES; kt += 2) {
        // phase A: compute tile kt from buf0; stage kt+1 -> buf1
        stageB(kt + 1, 1);
        loadA(a0p, kt + 1, a0n);
        loadA(a1p, kt + 1, a1n);
        mmaTile(Bs, a0c, a1c);
        __syncthreads();

        // phase B: compute tile kt+1 from buf1; stage kt+2 -> buf0
        if (kt + 2 < KTILES) {
            stageB(kt + 2, 0);
            loadA(a0p, kt + 2, a0c);
            loadA(a1p, kt + 2, a1c);
        }
        mmaTile(Bs + BS_TILE_USH, a0n, a1n);
        __syncthreads();
    }

    // C/D layout: lanes 0-15 -> rows 0..7 in vgpr 0..7, lanes 16-31 -> rows 8..15
    const int m      = lane & 15;
    const int rbase  = (lane >> 4) ? 8 : 0;
    const long long r0 = (long long)t0 * 16;
    const long long r1 = (long long)t1 * 16;
#pragma unroll
    for (int nt = 0; nt < 8; ++nt) {
        int col = nt * 16 + m;
        float bv = bias[col];
#pragma unroll
        for (int i = 0; i < 8; ++i) {
            long long ra = r0 + rbase + i;
            long long rb = r1 + rbase + i;
            float v0 = acc0[nt][i] + bv;
            float v1 = acc1[nt][i] + bv;
            if (ra < N) out[ra * D_ + col] = v0 > 0.0f ? v0 : 0.0f;
            if (rb < N) out[rb * D_ + col] = v1 > 0.0f ? v1 : 0.0f;
        }
    }
}

// ------------------------------ sigmoid head -------------------------------

__global__ void rgcn_head_kernel(const float* __restrict__ h,
                                 const float* __restrict__ w,
                                 const float* __restrict__ b,
                                 float* __restrict__ out, int N) {
    long long gid = (long long)blockIdx.x * blockDim.x + threadIdx.x;
    int node = (int)(gid >> 5);
    int lane = (int)(gid & 31);
    if (node >= N) return;
    const float4 hv = *(const float4*)(h + (long long)node * D_ + lane * 4);
    const float4 wv = *(const float4*)(w + lane * 4);
    float p = hv.x * wv.x + hv.y * wv.y + hv.z * wv.z + hv.w * wv.w;
#pragma unroll
    for (int off = 16; off > 0; off >>= 1) p += __shfl_xor(p, off, 32);
    if (lane == 0) out[node] = 1.0f / (1.0f + expf(-(p + b[0])));
}

// ---------------------------------------------------------------------------

extern "C" void kernel_launch(void* const* d_in, const int* in_sizes, int n_in,
                              void* d_out, int out_size, void* d_ws, size_t ws_size,
                              hipStream_t stream) {
    const float* x      = (const float*)d_in[0];
    const int*   ei     = (const int*)d_in[1];   // [2,E]
    const int*   et     = (const int*)d_in[2];   // [E]
    const float* Wrel1  = (const float*)d_in[3]; // [8,128,128] == [1024,128]
    const float* Wroot1 = (const float*)d_in[4]; // [128,128]
    const float* b1     = (const float*)d_in[5];
    const float* Wrel2  = (const float*)d_in[6];
    const float* Wroot2 = (const float*)d_in[7];
    const float* b2     = (const float*)d_in[8];
    const float* outw   = (const float*)d_in[9];
    const float* outb   = (const float*)d_in[10];
    float* out = (float*)d_out;

    const int N = in_sizes[0] / D_;   // 100000
    const int E = in_sizes[2];        // 1600000
    const int ntile16 = (N + 15) / 16;

    // ---- carve workspace ----
    char* p = (char*)d_ws;
    auto carve = [&](size_t bytes) -> void* {
        void* r = (void*)p;
        p += (bytes + 511) & ~(size_t)511;
        return r;
    };
    unsigned int*   cnt  = (unsigned int*)  carve((size_t)N * R_ * 4);
    float*          inv  = (float*)         carve((size_t)N * R_ * 4);
    float*          agg  = (float*)         carve((size_t)N * R_ * D_ * 4);
    unsigned short* Abuf = (unsigned short*)carve((size_t)ntile16 * ATILE_USH * 2);
    unsigned short* Bp   = (unsigned short*)carve((size_t)KTILES * 8 * 32 * 16 * 2);
    float*          h1   = (float*)         carve((size_t)N * D_ * 4);
    float*          h2   = (float*)         carve((size_t)N * D_ * 4);

    const int TPB = 256;
    dim3 blk(TPB);
    int g_cnt   = (E + TPB - 1) / TPB;
    int g_inv   = (N * R_ + TPB - 1) / TPB;
    int g_scat  = (int)(((long long)E * 32 + TPB - 1) / TPB);
    int g_build = (int)(((long long)N * 144 + TPB - 1) / TPB);
    int g_packb = (KTILES * 8 * 32 + TPB - 1) / TPB;
    int g_gemm  = (ntile16 + 15) / 16;     // 8 waves/block, 2 tiles (32 rows)/wave
    int g_head  = (int)(((long long)N * 32 + TPB - 1) / TPB);

    // Edge counts shared by both layers (same graph).
    hipMemsetAsync(cnt, 0, (size_t)N * R_ * 4, stream);
    rgcn_count_kernel<<<g_cnt, blk, 0, stream>>>(ei, et, cnt, E);
    rgcn_inv_kernel<<<g_inv, blk, 0, stream>>>(cnt, inv, N * R_);

    // ---- layer 1 ----
    hipMemsetAsync(agg, 0, (size_t)N * R_ * D_ * 4, stream);
    rgcn_scatter_kernel<<<g_scat, blk, 0, stream>>>(x, ei, et, agg, E);
    rgcn_build_a_kernel<<<g_build, blk, 0, stream>>>(agg, inv, x, Abuf, N);
    rgcn_pack_b_kernel<<<g_packb, blk, 0, stream>>>(Wrel1, Wroot1, Bp);
    rgcn_gemm_kernel<<<g_gemm, blk, 0, stream>>>(Abuf, Bp, b1, h1, N);

    // ---- layer 2 ----
    hipMemsetAsync(agg, 0, (size_t)N * R_ * D_ * 4, stream);
    rgcn_scatter_kernel<<<g_scat, blk, 0, stream>>>(h1, ei, et, agg, E);
    rgcn_build_a_kernel<<<g_build, blk, 0, stream>>>(agg, inv, h1, Abuf, N);
    rgcn_pack_b_kernel<<<g_packb, blk, 0, stream>>>(Wrel2, Wroot2, Bp);
    rgcn_gemm_kernel<<<g_gemm, blk, 0, stream>>>(Abuf, Bp, b2, h2, N);

    // ---- head ----
    rgcn_head_kernel<<<g_head, blk, 0, stream>>>(h2, outw, outb, out, N);
}
